// LmsPAM_Context_Module_38233798869170
// MI455X (gfx1250) — compile-verified, matrix operand order checked by reading, not run on hardware
//
#include <hip/hip_runtime.h>
#include <hip/hip_bf16.h>

// ---------------------------------------------------------------------------
// Linear-attention context module on MI455X (gfx1250, wave32, WMMA).
//
// Shapes: B=8, C=512, d=64, N=96*96=9216.
// ~111 GFLOP total, ~453MB compulsory traffic -> balanced at ~20-45us; all
// GEMMs routed through v_wmma_f32_16x16x32_bf16 (bf16 in, f32 accum).
//
// Restructure: attention[:,n] = M1@Q_n*norm_n + M2@Q_n*ynorm_n + bl, where
// M1 = Wl@KV^T, M2 = Wl@yKV^T (tiny per-batch 512x64 mats), which removes the
// need to materialize `w` and re-run a 512x512xN GEMM.
//
// kv_kernel uses the gfx1250 async-to-LDS path (GLOBAL_LOAD_ASYNC_TO_LDS_B128
// + S_WAIT_ASYNCCNT, double-buffered) when the builtins are available; it
// falls back to direct global-load fragments otherwise.
// ---------------------------------------------------------------------------

typedef __attribute__((ext_vector_type(16))) __bf16 v16bf;
typedef __attribute__((ext_vector_type(8)))  __bf16 v8bf;
typedef __attribute__((ext_vector_type(8)))  float  v8f;
typedef __attribute__((ext_vector_type(4)))  int    v4i;

constexpr int   Bb = 8, Cc = 512, Dd = 64, Nn = 9216;
constexpr float PPARAM = 10.0f;
constexpr float EEPS   = 1e-10f;

#if defined(__gfx1250__) && __has_builtin(__builtin_amdgcn_global_load_async_to_lds_b128)
#define USE_ASYNC_LDS 1
#else
#define USE_ASYNC_LDS 0
#endif

#if USE_ASYNC_LDS
// Signature (from this toolchain's diagnostics): both pointer params are
// NON-const int4* — AS(1) source, AS(3) dest — then imm offset, imm cpol.
#define GAS __attribute__((address_space(1)))
#define LAS __attribute__((address_space(3)))
#if __has_builtin(__builtin_amdgcn_s_wait_asynccnt)
#define WAIT_ASYNCCNT(n) __builtin_amdgcn_s_wait_asynccnt(n)
#else
#define WAIT_ASYNCCNT(n) asm volatile("s_wait_asynccnt %0" ::"i"(n) : "memory")
#endif
#endif

__device__ __forceinline__ v8f wmma_bf16(v16bf a, v16bf b, v8f c) {
  // 8 args: (neg_a, A, neg_b, B, c_mod, C, reuse_a, reuse_b)
  return __builtin_amdgcn_wmma_f32_16x16x32_bf16(false, a, false, b, (short)0, c,
                                                 false, false);
}

// Load one 16x32 bf16 WMMA operand fragment from a row-major [laneDim][kDim]
// buffer (global or LDS). ISA layout (CDNA5 ISA, 16-bit A-matrix 16x32):
// lane L: row = L&15, k elements = {half*8 + 0..7, 16 + half*8 + 0..7},
// half = L>>4. Two 16-byte loads per lane. The B operand uses the mirrored
// layout (lane = N column), so the same loader works when the buffer is
// stored [n][k].
__device__ __forceinline__ v16bf load_frag(const __bf16* base, int ld, int row0,
                                           int k0) {
  const int lane = threadIdx.x & 31;
  const int r = lane & 15, h = lane >> 4;
  const __bf16* p = base + (size_t)(row0 + r) * ld + k0 + h * 8;
  v8bf lo = *(const v8bf*)p;
  v8bf hi = *(const v8bf*)(p + 16);
  v16bf f;
#pragma unroll
  for (int i = 0; i < 8; ++i) { f[i] = lo[i]; f[i + 8] = hi[i]; }
  return f;
}

__device__ __forceinline__ void zero_acc(v8f& a) {
#pragma unroll
  for (int v = 0; v < 8; ++v) a[v] = 0.0f;
}

// elus_feature_map: p*relu(v) + exp(p*min(v,0)) - 1 + 1
//   v > 0 : 10*v + 1 ;  v <= 0 : exp(10*v)
__device__ __forceinline__ float elu_map(float v) {
  return v > 0.0f ? fmaf(PPARAM, v, 1.0f) : __expf(PPARAM * v);
}

// ---------------------------------------------------------------------------
// K0: weights -> bf16 (stacked [Wq;Wk1;Wv] = 640x512 and Wk2 = 64x512),
//     zero the f32 accumulators (KVT / yKVT / Ksum / yKsum).
// ---------------------------------------------------------------------------
__global__ __launch_bounds__(256) void prep_kernel(
    const float* __restrict__ Wq, const float* __restrict__ Wk1,
    const float* __restrict__ Wv, const float* __restrict__ Wk2,
    __bf16* __restrict__ Wxb, __bf16* __restrict__ Wyb,
    float* __restrict__ KVT, float* __restrict__ yKVT,
    float* __restrict__ Ksum, float* __restrict__ yKsum) {
  const int i = blockIdx.x * 256 + threadIdx.x;
  if (i < 640 * 512) {
    const int r = i >> 9, c = i & 511;
    float v = (r < 64)    ? Wq[r * 512 + c]
              : (r < 128) ? Wk1[(r - 64) * 512 + c]
                          : Wv[(r - 128) * 512 + c];
    Wxb[i] = (__bf16)v;
  }
  if (i < 64 * 512) Wyb[i] = (__bf16)Wk2[i];
  if (i < Bb * Cc * Dd) { KVT[i] = 0.0f; yKVT[i] = 0.0f; }
  if (i < Bb * Dd) { Ksum[i] = 0.0f; yKsum[i] = 0.0f; }
}

// ---------------------------------------------------------------------------
// K1: projections. grid = (N/64, 6, B), 256 threads = 8 waves.
// Row-blocks 0..4: 128 rows of [Wq;Wk1;Wv] @ x ; row-block 5: 64 rows Wk2 @ y.
// x/y chunk is staged transposed in LDS as bf16 ([n][c], two 256-deep k
// phases) so the B-operand fragments are k-contiguous per lane. (Staging must
// stay on the VGPR path: it does f32->bf16 conversion + transpose, which the
// async/TDM byte-copy engines cannot do.)
// Epilogue fuses bias + ELU feature map; outputs stored bf16 [row][N].
// ---------------------------------------------------------------------------
__global__ __launch_bounds__(256) void proj_kernel(
    const float* __restrict__ xin, const float* __restrict__ yin,
    const __bf16* __restrict__ Wxb, const __bf16* __restrict__ Wyb,
    const float* __restrict__ bq, const float* __restrict__ bk1,
    const float* __restrict__ bk2, const float* __restrict__ bv,
    __bf16* __restrict__ Qb, __bf16* __restrict__ Kb,
    __bf16* __restrict__ yKb, __bf16* __restrict__ Vb) {
  __shared__ __align__(16) __bf16 sB[64 * 264];  // [n=64][c=256 (+8 pad)]

  const int nbase = blockIdx.x * 64;
  const int rb    = blockIdx.y;   // 0..5
  const int b     = blockIdx.z;
  const int wid   = threadIdx.x >> 5;
  const int wr    = wid & 1;      // row half of the 128-row block
  const int wc    = wid >> 1;     // 16-col strip
  const bool active = (rb < 5) || (wr == 0);
  const float* src = (rb == 5) ? yin : xin;

  v8f acc[4];
#pragma unroll
  for (int t = 0; t < 4; ++t) zero_acc(acc[t]);

  for (int ph = 0; ph < 2; ++ph) {
    __syncthreads();
    // load 256x64 f32 chunk, convert to bf16, store transposed [n][c]
#pragma unroll 4
    for (int j = 0; j < 64; ++j) {
      const int i = threadIdx.x + 256 * j;
      const int n = i & 63, c = i >> 6;
      float v = src[((size_t)b * Cc + ph * 256 + c) * Nn + nbase + n];
      sB[n * 264 + c] = (__bf16)v;
    }
    __syncthreads();
    if (active) {
      for (int kk = 0; kk < 8; ++kk) {
        const int kg = ph * 256 + kk * 32;
        v16bf bf = load_frag(sB, 264, wc * 16, kk * 32);
#pragma unroll
        for (int t = 0; t < 4; ++t) {
          const __bf16* Ab;
          int rowg;
          if (rb < 5) { Ab = Wxb; rowg = rb * 128 + wr * 64 + t * 16; }
          else        { Ab = Wyb; rowg = t * 16; }
          v16bf af = load_frag(Ab, 512, rowg, kg);
          acc[t] = wmma_bf16(af, bf, acc[t]);
        }
      }
    }
  }
  if (!active) return;

  const int lane = threadIdx.x & 31, cl = lane & 15, hf = lane >> 4;
  const int n = nbase + wc * 16 + cl;
#pragma unroll
  for (int t = 0; t < 4; ++t) {
#pragma unroll
    for (int v = 0; v < 8; ++v) {
      const int rloc = t * 16 + v + hf * 8;  // 0..63 within this wave's half
      float val = acc[t][v];
      if (rb == 0) {
        if (wr == 0) {  // Q rows 0..63
          val = elu_map(val + bq[rloc]);
          Qb[((size_t)b * Dd + rloc) * Nn + n] = (__bf16)val;
        } else {        // K rows 64..127
          val = elu_map(val + bk1[rloc]);
          Kb[((size_t)b * Dd + rloc) * Nn + n] = (__bf16)val;
        }
      } else if (rb < 5) {  // V rows
        const int vrow = (rb - 1) * 128 + wr * 64 + rloc;
        val += bv[vrow];
        Vb[((size_t)b * Cc + vrow) * Nn + n] = (__bf16)val;
      } else {            // yK rows
        val = elu_map(val + bk2[rloc]);
        yKb[((size_t)b * Dd + rloc) * Nn + n] = (__bf16)val;
      }
    }
  }
}

// ---------------------------------------------------------------------------
// K2: row sums of K / yK (for the norms). One block per (which,b,m) row.
// ---------------------------------------------------------------------------
__global__ __launch_bounds__(256) void ksum_kernel(
    const __bf16* __restrict__ Kb, const __bf16* __restrict__ yKb,
    float* __restrict__ Ksum, float* __restrict__ yKsum) {
  __shared__ float red[256];
  const int row = blockIdx.x;       // 0..1023
  const int which = row >> 9;       // 0 = K, 1 = yK
  const int bm = row & 511;         // flat (b*64+m)
  const __bf16* src = (which ? yKb : Kb) + (size_t)bm * Nn;
  float s = 0.0f;
  for (int i = threadIdx.x; i < Nn; i += 256) s += (float)src[i];
  red[threadIdx.x] = s;
  __syncthreads();
  for (int off = 128; off > 0; off >>= 1) {
    if (threadIdx.x < (unsigned)off) red[threadIdx.x] += red[threadIdx.x + off];
    __syncthreads();
  }
  if (threadIdx.x == 0) (which ? yKsum : Ksum)[bm] = red[0];
}

// ---------------------------------------------------------------------------
// K3: KVT[b][c][m] = sum_n V[c][n]*K[m][n] and yKVT with yK.
// A-operand (V rows, per-wave private) streams straight from global.
// B-operands (K / yK, shared by all waves) are staged into padded LDS tiles
// with async DMA (GLOBAL_LOAD_ASYNC_TO_LDS_B128), double-buffered: issue
// chunk c+1, s_wait_asynccnt 4 (in-order retirement => chunk c is resident),
// barrier, compute chunk c. Falls back to direct global fragments if the
// async builtins are unavailable.
// grid = (8 nchunks, 4 cblocks, B), 256 threads; N chunk = 1152 per block.
// ---------------------------------------------------------------------------
__global__ __launch_bounds__(256) void kv_kernel(
    const __bf16* __restrict__ Kb, const __bf16* __restrict__ yKb,
    const __bf16* __restrict__ Vb, float* __restrict__ KVT,
    float* __restrict__ yKVT) {
  const int nc = blockIdx.x, cb = blockIdx.y, b = blockIdx.z;
  const int wid = threadIdx.x >> 5;
  const __bf16* Vbase = Vb  + (size_t)b * Cc * Nn;
  const __bf16* Kbase = Kb  + (size_t)b * Dd * Nn;
  const __bf16* Ybase = yKb + (size_t)b * Dd * Nn;
  const int crow0 = cb * 128 + wid * 16;
  const int nbase = nc * 1152;

  v8f aK[4], aY[4];
#pragma unroll
  for (int t = 0; t < 4; ++t) { zero_acc(aK[t]); zero_acc(aY[t]); }

#if USE_ASYNC_LDS
  // [buf][row 0..63][k 0..63 (+8 pad)], 16B async granules land on the padded
  // row stride (72*2B = 144B, 16B-aligned).
  __shared__ __align__(16) __bf16 sK[2][64 * 72];
  __shared__ __align__(16) __bf16 sY[2][64 * 72];

  auto issue_chunk = [&](int ch, int bf) {
    const int k0 = nbase + ch * 64;
#pragma unroll
    for (int j = 0; j < 2; ++j) {
      const int i = (int)threadIdx.x + 256 * j;  // 0..511 granules per matrix
      const int row = i >> 3, gg = i & 7;
      const __bf16* gK = Kbase + (size_t)row * Nn + k0 + gg * 8;
      const __bf16* gY = Ybase + (size_t)row * Nn + k0 + gg * 8;
      __builtin_amdgcn_global_load_async_to_lds_b128(
          (GAS v4i*)(const void*)gK,
          (LAS v4i*)(void*)&sK[bf][row * 72 + gg * 8], 0, 0);
      __builtin_amdgcn_global_load_async_to_lds_b128(
          (GAS v4i*)(const void*)gY,
          (LAS v4i*)(void*)&sY[bf][row * 72 + gg * 8], 0, 0);
    }
  };

  int buf = 0;
  issue_chunk(0, 0);
  for (int ch = 0; ch < 18; ++ch) {  // 18 chunks of 64 n
    if (ch + 1 < 18) {
      issue_chunk(ch + 1, buf ^ 1);
      WAIT_ASYNCCNT(4);  // 4 newer ops in flight -> the 4 for chunk ch are done
    } else {
      WAIT_ASYNCCNT(0);
    }
    __syncthreads();
#pragma unroll
    for (int kk = 0; kk < 2; ++kk) {
      const int k0 = nbase + ch * 64 + kk * 32;
      v16bf af = load_frag(Vbase, Nn, crow0, k0);      // A: V rows (M=c)
#pragma unroll
      for (int t = 0; t < 4; ++t) {
        v16bf bk = load_frag(sK[buf], 72, t * 16, kk * 32);  // B: lane=m, k=n
        aK[t] = wmma_bf16(af, bk, aK[t]);
        v16bf by = load_frag(sY[buf], 72, t * 16, kk * 32);
        aY[t] = wmma_bf16(af, by, aY[t]);
      }
    }
    __syncthreads();  // everyone done reading `buf` before it is re-filled
    buf ^= 1;
  }
#else
  for (int it = 0; it < 36; ++it) {
    const int k0 = nbase + it * 32;
    __builtin_prefetch(Vbase + (size_t)crow0 * Nn + k0 + 512, 0, 1);
    v16bf af = load_frag(Vbase, Nn, crow0, k0);
#pragma unroll
    for (int t = 0; t < 4; ++t) {
      v16bf bk = load_frag(Kbase, Nn, t * 16, k0);
      aK[t] = wmma_bf16(af, bk, aK[t]);
      v16bf by = load_frag(Ybase, Nn, t * 16, k0);
      aY[t] = wmma_bf16(af, by, aY[t]);
    }
  }
#endif

  const int lane = threadIdx.x & 31, cl = lane & 15, hf = lane >> 4;
#pragma unroll
  for (int t = 0; t < 4; ++t) {
#pragma unroll
    for (int v = 0; v < 8; ++v) {
      const int c = crow0 + v + hf * 8;
      const int m = t * 16 + cl;
      atomicAdd(&KVT [((size_t)b * Cc + c) * Dd + m], aK[t][v]);
      atomicAdd(&yKVT[((size_t)b * Cc + c) * Dd + m], aY[t][v]);
    }
  }
}

// ---------------------------------------------------------------------------
// K4: M1 = Wl @ KV^T, M2 = Wl @ yKV^T (537 MFLOP total -> scalar is fine).
// Output stored bf16 [b][o][m] (m-contiguous => direct A-fragments in K6).
// ---------------------------------------------------------------------------
__global__ __launch_bounds__(256) void m_kernel(
    const float* __restrict__ Wl, const float* __restrict__ KVT,
    const float* __restrict__ yKVT, __bf16* __restrict__ M1,
    __bf16* __restrict__ M2) {
  const int idx = blockIdx.x * 256 + threadIdx.x;  // (b*512+o)*64+m
  const int m = idx & 63, o = (idx >> 6) & 511, b = idx >> 15;
  const float* wrow = Wl + (size_t)o * Cc;
  const float* kcol = KVT  + (size_t)b * Cc * Dd + m;
  const float* ycol = yKVT + (size_t)b * Cc * Dd + m;
  float s1 = 0.0f, s2 = 0.0f;
#pragma unroll 4
  for (int c = 0; c < Cc; ++c) {
    const float w = wrow[c];
    s1 = fmaf(w, kcol[(size_t)c * Dd], s1);
    s2 = fmaf(w, ycol[(size_t)c * Dd], s2);
  }
  M1[idx] = (__bf16)s1;
  M2[idx] = (__bf16)s2;
}

// ---------------------------------------------------------------------------
// K5: per-pixel norms: 1 / (Q_n . (Ksum + EPS)), same for yKsum.
// ---------------------------------------------------------------------------
__global__ __launch_bounds__(256) void norm_kernel(
    const __bf16* __restrict__ Qb, const float* __restrict__ Ksum,
    const float* __restrict__ yKsum, float* __restrict__ normv,
    float* __restrict__ ynormv) {
  const int idx = blockIdx.x * 256 + threadIdx.x;
  if (idx >= Bb * Nn) return;
  const int b = idx / Nn, n = idx - b * Nn;
  const __bf16* q = Qb + (size_t)b * Dd * Nn + n;
  const float* ks = Ksum + b * Dd;
  const float* ys = yKsum + b * Dd;
  float d1 = 0.0f, d2 = 0.0f;
#pragma unroll 8
  for (int m = 0; m < Dd; ++m) {
    const float qv = (float)q[(size_t)m * Nn];
    d1 = fmaf(qv, ks[m] + EEPS, d1);
    d2 = fmaf(qv, ys[m] + EEPS, d2);
  }
  normv[idx]  = 1.0f / d1;
  ynormv[idx] = 1.0f / d2;
}

// ---------------------------------------------------------------------------
// K6: out[:,n] = x[:,n] + gamma*(norm_n*(M1@Q_n) + ynorm_n*(M2@Q_n) + bl).
// grid = (N/32, 2, B), 256 threads = 8 waves, each wave 32 rows x 32 cols.
// Q tile transposed via LDS so B-fragments are k(=m)-contiguous.
// ---------------------------------------------------------------------------
__global__ __launch_bounds__(256) void final_kernel(
    const float* __restrict__ xin, const __bf16* __restrict__ Qb,
    const __bf16* __restrict__ M1, const __bf16* __restrict__ M2,
    const float* __restrict__ normv, const float* __restrict__ ynormv,
    const float* __restrict__ bl, const float* __restrict__ gamma,
    float* __restrict__ outp) {
  __shared__ __align__(16) __bf16 sQ[32 * 72];  // [n=32][m=64 (+8 pad)]
  const int n0   = blockIdx.x * 32;
  const int rblk = blockIdx.y;
  const int b    = blockIdx.z;

#pragma unroll
  for (int j = 0; j < 8; ++j) {
    const int i = threadIdx.x + 256 * j;  // 2048 bf16 elements
    const int n = i & 31, m = i >> 5;
    sQ[n * 72 + m] = Qb[((size_t)b * Dd + m) * Nn + n0 + n];
  }
  __syncthreads();

  const int wid = threadIdx.x >> 5;
  const int r0  = rblk * 256 + wid * 32;
  const __bf16* M1b = M1 + (size_t)b * Cc * Dd;
  const __bf16* M2b = M2 + (size_t)b * Cc * Dd;

  v8f a1[2][2], a2[2][2];
#pragma unroll
  for (int tr = 0; tr < 2; ++tr)
#pragma unroll
    for (int tc = 0; tc < 2; ++tc) { zero_acc(a1[tr][tc]); zero_acc(a2[tr][tc]); }

#pragma unroll
  for (int kk = 0; kk < 2; ++kk) {
    v16bf bf0 = load_frag(sQ, 72, 0, kk * 32);
    v16bf bf1 = load_frag(sQ, 72, 16, kk * 32);
#pragma unroll
    for (int tr = 0; tr < 2; ++tr) {
      v16bf f1 = load_frag(M1b, 64, r0 + tr * 16, kk * 32);
      v16bf f2 = load_frag(M2b, 64, r0 + tr * 16, kk * 32);
      a1[tr][0] = wmma_bf16(f1, bf0, a1[tr][0]);
      a1[tr][1] = wmma_bf16(f1, bf1, a1[tr][1]);
      a2[tr][0] = wmma_bf16(f2, bf0, a2[tr][0]);
      a2[tr][1] = wmma_bf16(f2, bf1, a2[tr][1]);
    }
  }

  const float g = gamma[0];
  const int lane = threadIdx.x & 31, cl = lane & 15, hf = lane >> 4;
#pragma unroll
  for (int tc = 0; tc < 2; ++tc) {
    const int n = n0 + tc * 16 + cl;
    const float s1 = normv[b * Nn + n];
    const float s2 = ynormv[b * Nn + n];
#pragma unroll
    for (int tr = 0; tr < 2; ++tr) {
#pragma unroll
      for (int v = 0; v < 8; ++v) {
        const int R = r0 + tr * 16 + v + hf * 8;
        const float att = fmaf(s1, a1[tr][tc][v], fmaf(s2, a2[tr][tc][v], bl[R]));
        const size_t idx = ((size_t)b * Cc + R) * Nn + n;
        outp[idx] = fmaf(g, att, xin[idx]);
      }
    }
  }
}

// ---------------------------------------------------------------------------
extern "C" void kernel_launch(void* const* d_in, const int* in_sizes, int n_in,
                              void* d_out, int out_size, void* d_ws, size_t ws_size,
                              hipStream_t stream) {
  const float* x     = (const float*)d_in[0];
  const float* y     = (const float*)d_in[1];
  const float* Wq    = (const float*)d_in[2];
  const float* bq    = (const float*)d_in[3];
  const float* Wk1   = (const float*)d_in[4];
  const float* bk1   = (const float*)d_in[5];
  const float* Wk2   = (const float*)d_in[6];
  const float* bk2   = (const float*)d_in[7];
  const float* Wv    = (const float*)d_in[8];
  const float* bv    = (const float*)d_in[9];
  const float* Wl    = (const float*)d_in[10];
  const float* bl    = (const float*)d_in[11];
  const float* gamma = (const float*)d_in[12];
  float* outp = (float*)d_out;
  (void)in_sizes; (void)n_in; (void)out_size; (void)ws_size;

  // Workspace carve (~110 MB total), 256B aligned regions.
  char* p = (char*)d_ws;
  auto carve = [&](size_t bytes) -> char* {
    char* r = p;
    p += (bytes + 255) & ~(size_t)255;
    return r;
  };
  __bf16* Wxb   = (__bf16*)carve((size_t)640 * 512 * 2);
  __bf16* Wyb   = (__bf16*)carve((size_t)64 * 512 * 2);
  __bf16* Qb    = (__bf16*)carve((size_t)Bb * Dd * Nn * 2);
  __bf16* Kb    = (__bf16*)carve((size_t)Bb * Dd * Nn * 2);
  __bf16* yKb   = (__bf16*)carve((size_t)Bb * Dd * Nn * 2);
  __bf16* Vb    = (__bf16*)carve((size_t)Bb * Cc * Nn * 2);
  float* KVT    = (float*)carve((size_t)Bb * Cc * Dd * 4);
  float* yKVT   = (float*)carve((size_t)Bb * Cc * Dd * 4);
  float* Ksum   = (float*)carve((size_t)Bb * Dd * 4);
  float* yKsum  = (float*)carve((size_t)Bb * Dd * 4);
  __bf16* M1    = (__bf16*)carve((size_t)Bb * Cc * Dd * 2);
  __bf16* M2    = (__bf16*)carve((size_t)Bb * Cc * Dd * 2);
  float* normv  = (float*)carve((size_t)Bb * Nn * 4);
  float* ynormv = (float*)carve((size_t)Bb * Nn * 4);

  prep_kernel<<<dim3((640 * 512 + 255) / 256), 256, 0, stream>>>(
      Wq, Wk1, Wv, Wk2, Wxb, Wyb, KVT, yKVT, Ksum, yKsum);
  proj_kernel<<<dim3(Nn / 64, 6, Bb), 256, 0, stream>>>(
      x, y, Wxb, Wyb, bq, bk1, bk2, bv, Qb, Kb, yKb, Vb);
  ksum_kernel<<<dim3(2 * Bb * Dd), 256, 0, stream>>>(Kb, yKb, Ksum, yKsum);
  kv_kernel<<<dim3(8, 4, Bb), 256, 0, stream>>>(Kb, yKb, Vb, KVT, yKVT);
  m_kernel<<<dim3((Bb * Cc * Dd + 255) / 256), 256, 0, stream>>>(
      Wl, KVT, yKVT, M1, M2);
  norm_kernel<<<dim3((Bb * Nn + 255) / 256), 256, 0, stream>>>(
      Qb, Ksum, yKsum, normv, ynormv);
  final_kernel<<<dim3(Nn / 32, 2, Bb), 256, 0, stream>>>(
      x, Qb, M1, M2, normv, ynormv, bl, gamma, outp);
}
